// MultiheadSelfAttention_37572373906085
// MI455X (gfx1250) — compile-verified, matrix operand order checked by reading, not run on hardware
//
#include <hip/hip_runtime.h>
#include <hip/hip_bf16.h>
#include <math.h>

typedef __bf16 bf16_t;
typedef __attribute__((ext_vector_type(16))) __bf16 v16bf;
typedef __attribute__((ext_vector_type(8)))  __bf16 v8bf;
typedef __attribute__((ext_vector_type(8)))  float  v8f;

#define D_MODEL 1024
#define N_HEAD  16
#define HEAD    64
#define SEQ     2048
#define BATCH   4
#define NROW    (BATCH*SEQ)   /* 8192 */
#define SCALE   0.125f        /* 1/sqrt(64) */
#define PPITCH  72            /* LDS pitch for P tile (bf16 elems), 144B = 16B-aligned */

__device__ __forceinline__ v8f zero8() {
  v8f z;
#pragma unroll
  for (int i = 0; i < 8; ++i) z[i] = 0.f;
  return z;
}

// Load one 16-element bf16 WMMA fragment half-row: elements [0..7] and [16..23]
// relative to p.  Caller applies the (lane>=16 ? +8 : 0) K offset.
__device__ __forceinline__ v16bf load_frag(const bf16_t* p) {
  v8bf lo = *(const v8bf*)(p);
  v8bf hi = *(const v8bf*)(p + 16);
  v16bf r;
#pragma unroll
  for (int i = 0; i < 8; ++i) { r[i] = lo[i]; r[i + 8] = hi[i]; }
  return r;
}

__device__ __forceinline__ v8f wmma_bf16(v16bf a, v16bf b, v8f c) {
  return __builtin_amdgcn_wmma_f32_16x16x32_bf16(false, a, false, b, (short)0, c,
                                                 false, false);
}

// ---------------------------------------------------------------- conversions
__global__ void cvt_bf16_kernel(const float* __restrict__ in,
                                bf16_t* __restrict__ out, int n) {
  int i = blockIdx.x * blockDim.x + threadIdx.x;
  if (i < n) out[i] = (bf16_t)in[i];
}

// in: row-major [K,N] fp32  ->  out: row-major [N,K] bf16 (N-major / B^T form)
__global__ void tcvt_kernel(const float* __restrict__ in,
                            bf16_t* __restrict__ out, int K, int N) {
  int i = blockIdx.x * blockDim.x + threadIdx.x;
  if (i < N * K) {
    int n = i / K;
    int k = i - n * K;
    out[i] = (bf16_t)in[(size_t)k * N + n];
  }
}

// ---------------------------------------------------------------- QKV GEMM
// C[8192,3072] = Xb[8192,1024] * W_attn + b_attn, scattered into
// Q[B,H,S,64], K[B,H,S,64], Vt[B,H,64,S] as bf16.
__global__ __launch_bounds__(256)
void qkv_gemm_kernel(const bf16_t* __restrict__ A,    // [8192,1024]
                     const bf16_t* __restrict__ Bt,   // [3072,1024] N-major
                     const float*  __restrict__ bias, // [3072]
                     bf16_t* __restrict__ Qo,
                     bf16_t* __restrict__ Ko,
                     bf16_t* __restrict__ Vt) {
  const int tid  = threadIdx.x;
  const int lane = tid & 31;
  const int wave = tid >> 5;
  const int l16  = lane & 15;
  const int hi   = (lane >> 4) & 1;
  const int koff = hi * 8;
  const int wm = wave >> 1;                 // 0..3
  const int wn = wave & 1;                  // 0..1
  const int mbase = blockIdx.y * 128 + wm * 32;
  const int nbase = blockIdx.x * 128 + wn * 64;

  v8f acc[2][4];
#pragma unroll
  for (int i = 0; i < 2; ++i)
#pragma unroll
    for (int f = 0; f < 4; ++f) acc[i][f] = zero8();

#pragma unroll 4
  for (int kt = 0; kt < D_MODEL; kt += 32) {
    v16bf a0 = load_frag(A + (size_t)(mbase + l16) * D_MODEL + kt + koff);
    v16bf a1 = load_frag(A + (size_t)(mbase + 16 + l16) * D_MODEL + kt + koff);
#pragma unroll
    for (int f = 0; f < 4; ++f) {
      v16bf bfrag = load_frag(Bt + (size_t)(nbase + f * 16 + l16) * D_MODEL + kt + koff);
      acc[0][f] = wmma_bf16(a0, bfrag, acc[0][f]);
      acc[1][f] = wmma_bf16(a1, bfrag, acc[1][f]);
    }
  }

#pragma unroll
  for (int f = 0; f < 4; ++f) {
    const int nc = nbase + f * 16 + l16;    // global column, 0..3071
    const float bv = bias[nc];
    const int sector = nc >> 10;            // 0=Q 1=K 2=V (wave-uniform)
    const int dm = nc & 1023;
    const int h = dm >> 6;
    const int d = dm & 63;
#pragma unroll
    for (int i = 0; i < 2; ++i) {
#pragma unroll
      for (int r = 0; r < 8; ++r) {
        const int m = mbase + i * 16 + r + hi * 8;     // global row
        const int bb = m >> 11;                        // / 2048
        const int s  = m & 2047;
        const bf16_t val = (bf16_t)(acc[i][f][r] + bv);
        if (sector == 0)
          Qo[(((size_t)(bb * N_HEAD + h)) * SEQ + s) * HEAD + d] = val;
        else if (sector == 1)
          Ko[(((size_t)(bb * N_HEAD + h)) * SEQ + s) * HEAD + d] = val;
        else
          Vt[(((size_t)(bb * N_HEAD + h)) * HEAD + d) * SEQ + s] = val;
      }
    }
  }
}

// ---------------------------------------------------------------- attention
// Flash attention, causal. One block = 64 query rows of one (b,h);
// 4 waves x 16-row stripes. Key-tile loop of width 64.
__global__ __launch_bounds__(128)
void attn_kernel(const bf16_t* __restrict__ Q,    // [B,H,S,64]
                 const bf16_t* __restrict__ Kq,   // [B,H,S,64]
                 const bf16_t* __restrict__ Vt,   // [B,H,64,S]
                 bf16_t* __restrict__ Y) {        // [8192,1024] bf16
  __shared__ __align__(16) bf16_t lds_p[4 * 16 * PPITCH];

  const int tid  = threadIdx.x;
  const int lane = tid & 31;
  const int wave = tid >> 5;                // 0..3
  const int l16  = lane & 15;
  const int hi   = (lane >> 4) & 1;
  const int koff = hi * 8;

  const int bh = blockIdx.y;                // 0..63
  const int b  = bh >> 4;
  const int h  = bh & 15;
  const int qb = blockIdx.x * 64;

  const bf16_t* Qb = Q  + (size_t)bh * SEQ * HEAD;
  const bf16_t* Kb = Kq + (size_t)bh * SEQ * HEAD;
  const bf16_t* Vb = Vt + (size_t)bh * HEAD * SEQ;

  const int qrow_base = qb + wave * 16;

  v16bf qf[2];
#pragma unroll
  for (int ks = 0; ks < 2; ++ks)
    qf[ks] = load_frag(Qb + (size_t)(qrow_base + l16) * HEAD + ks * 32 + koff);

  v8f o[4];
  float runm[8], runl[8];
#pragma unroll
  for (int f = 0; f < 4; ++f) o[f] = zero8();
#pragma unroll
  for (int r = 0; r < 8; ++r) { runm[r] = -INFINITY; runl[r] = 0.f; }

  bf16_t* lp = &lds_p[wave * 16 * PPITCH];

  for (int kb = 0; kb <= qb; kb += 64) {
    // ---- scores S = Q(16x64) * K^T(64x64), 2 k-steps over head dim
    v8f sc[4];
#pragma unroll
    for (int f = 0; f < 4; ++f) {
      v8f a = zero8();
#pragma unroll
      for (int ks = 0; ks < 2; ++ks) {
        v16bf kf = load_frag(Kb + (size_t)(kb + f * 16 + l16) * HEAD + ks * 32 + koff);
        a = wmma_bf16(qf[ks], kf, a);
      }
      sc[f] = a;
    }
    // ---- scale + causal mask + per-row tile max
    float tmax[8];
#pragma unroll
    for (int r = 0; r < 8; ++r) tmax[r] = -INFINITY;
#pragma unroll
    for (int f = 0; f < 4; ++f) {
      const int key = kb + f * 16 + l16;
#pragma unroll
      for (int r = 0; r < 8; ++r) {
        const int qr = qrow_base + r + hi * 8;
        float s = sc[f][r] * SCALE;
        s = (key > qr) ? -INFINITY : s;   // select, keeps EXEC uniform
        sc[f][r] = s;
        tmax[r] = fmaxf(tmax[r], s);
      }
    }
#pragma unroll
    for (int r = 0; r < 8; ++r)
#pragma unroll
      for (int mm = 1; mm < 16; mm <<= 1)
        tmax[r] = fmaxf(tmax[r], __shfl_xor(tmax[r], mm, 32));

    float corr[8], tsum[8];
#pragma unroll
    for (int r = 0; r < 8; ++r) {
      const float nm = fmaxf(runm[r], tmax[r]);
      corr[r] = __expf(runm[r] - nm);
      runm[r] = nm;
      tsum[r] = 0.f;
    }
    // ---- exp, row-sum, spill P tile (C-layout) to LDS as bf16
#pragma unroll
    for (int f = 0; f < 4; ++f) {
#pragma unroll
      for (int r = 0; r < 8; ++r) {
        const float p = __expf(sc[f][r] - runm[r]);
        tsum[r] += p;
        lp[(r + hi * 8) * PPITCH + f * 16 + l16] = (bf16_t)p;
      }
    }
#pragma unroll
    for (int r = 0; r < 8; ++r) {
#pragma unroll
      for (int mm = 1; mm < 16; mm <<= 1)
        tsum[r] += __shfl_xor(tsum[r], mm, 32);
      runl[r] = runl[r] * corr[r] + tsum[r];
    }
    __syncthreads();
    // ---- reload P in A-fragment layout
    v16bf pf[2];
#pragma unroll
    for (int ks = 0; ks < 2; ++ks)
      pf[ks] = load_frag(lp + l16 * PPITCH + ks * 32 + koff);
    __syncthreads();
    // ---- O = O*corr + P * V   (V^T is N-major: contiguous along keys)
#pragma unroll
    for (int f = 0; f < 4; ++f) {
#pragma unroll
      for (int r = 0; r < 8; ++r) o[f][r] *= corr[r];
#pragma unroll
      for (int ks = 0; ks < 2; ++ks) {
        v16bf vf = load_frag(Vb + (size_t)(f * 16 + l16) * SEQ + kb + ks * 32 + koff);
        o[f] = wmma_bf16(pf[ks], vf, o[f]);
      }
    }
  }
  // ---- normalize and write Y (bf16, [b*S+q, h*64+d])
#pragma unroll
  for (int f = 0; f < 4; ++f) {
#pragma unroll
    for (int r = 0; r < 8; ++r) {
      const int qr = qrow_base + r + hi * 8;
      const float y = o[f][r] / runl[r];
      Y[((size_t)(b * SEQ + qr)) * D_MODEL + h * HEAD + f * 16 + l16] = (bf16_t)y;
    }
  }
}

// ---------------------------------------------------------------- proj GEMM
__global__ __launch_bounds__(256)
void proj_gemm_kernel(const bf16_t* __restrict__ A,    // [8192,1024] Yb
                      const bf16_t* __restrict__ Bt,   // [1024,1024] N-major
                      const float*  __restrict__ bias, // [1024]
                      float* __restrict__ out) {       // [8192,1024] fp32
  const int tid  = threadIdx.x;
  const int lane = tid & 31;
  const int wave = tid >> 5;
  const int l16  = lane & 15;
  const int hi   = (lane >> 4) & 1;
  const int koff = hi * 8;
  const int wm = wave >> 1;
  const int wn = wave & 1;
  const int mbase = blockIdx.y * 128 + wm * 32;
  const int nbase = blockIdx.x * 128 + wn * 64;

  v8f acc[2][4];
#pragma unroll
  for (int i = 0; i < 2; ++i)
#pragma unroll
    for (int f = 0; f < 4; ++f) acc[i][f] = zero8();

#pragma unroll 4
  for (int kt = 0; kt < D_MODEL; kt += 32) {
    v16bf a0 = load_frag(A + (size_t)(mbase + l16) * D_MODEL + kt + koff);
    v16bf a1 = load_frag(A + (size_t)(mbase + 16 + l16) * D_MODEL + kt + koff);
#pragma unroll
    for (int f = 0; f < 4; ++f) {
      v16bf bfrag = load_frag(Bt + (size_t)(nbase + f * 16 + l16) * D_MODEL + kt + koff);
      acc[0][f] = wmma_bf16(a0, bfrag, acc[0][f]);
      acc[1][f] = wmma_bf16(a1, bfrag, acc[1][f]);
    }
  }

#pragma unroll
  for (int f = 0; f < 4; ++f) {
    const int nc = nbase + f * 16 + l16;
    const float bv = bias[nc];
#pragma unroll
    for (int i = 0; i < 2; ++i) {
#pragma unroll
      for (int r = 0; r < 8; ++r) {
        const int m = mbase + i * 16 + r + hi * 8;
        out[(size_t)m * D_MODEL + nc] = acc[i][f][r] + bv;
      }
    }
  }
}

// ---------------------------------------------------------------- launcher
extern "C" void kernel_launch(void* const* d_in, const int* in_sizes, int n_in,
                              void* d_out, int out_size, void* d_ws, size_t ws_size,
                              hipStream_t stream) {
  const float* X  = (const float*)d_in[0];   // [4,2048,1024]
  const float* Wa = (const float*)d_in[1];   // [1024,3072]
  const float* ba = (const float*)d_in[2];   // [3072]
  const float* Wp = (const float*)d_in[3];   // [1024,1024]
  const float* bp = (const float*)d_in[4];   // [1024]
  float* out = (float*)d_out;

  char* ws = (char*)d_ws;
  size_t off = 0;
  bf16_t* Xb  = (bf16_t*)(ws + off); off += (size_t)NROW * D_MODEL * 2;      // 16 MB
  bf16_t* WaT = (bf16_t*)(ws + off); off += (size_t)3 * D_MODEL * D_MODEL * 2; // 6 MB
  bf16_t* WpT = (bf16_t*)(ws + off); off += (size_t)D_MODEL * D_MODEL * 2;   // 2 MB
  bf16_t* Qd  = (bf16_t*)(ws + off); off += (size_t)NROW * D_MODEL * 2;      // 16 MB
  bf16_t* Kd  = (bf16_t*)(ws + off); off += (size_t)NROW * D_MODEL * 2;      // 16 MB
  bf16_t* Vtd = (bf16_t*)(ws + off); off += (size_t)NROW * D_MODEL * 2;      // 16 MB
  bf16_t* Yb  = (bf16_t*)(ws + off); off += (size_t)NROW * D_MODEL * 2;      // 16 MB

  {
    int n = NROW * D_MODEL;
    cvt_bf16_kernel<<<(n + 255) / 256, 256, 0, stream>>>(X, Xb, n);
  }
  {
    int n = 3 * D_MODEL * D_MODEL;
    tcvt_kernel<<<(n + 255) / 256, 256, 0, stream>>>(Wa, WaT, D_MODEL, 3 * D_MODEL);
  }
  {
    int n = D_MODEL * D_MODEL;
    tcvt_kernel<<<(n + 255) / 256, 256, 0, stream>>>(Wp, WpT, D_MODEL, D_MODEL);
  }
  qkv_gemm_kernel<<<dim3(3 * D_MODEL / 128, NROW / 128), 256, 0, stream>>>(
      Xb, WaT, ba, Qd, Kd, Vtd);
  attn_kernel<<<dim3(SEQ / 64, BATCH * N_HEAD), 128, 0, stream>>>(Qd, Kd, Vtd, Yb);
  proj_gemm_kernel<<<dim3(D_MODEL / 128, NROW / 128), 256, 0, stream>>>(
      Yb, WpT, bp, out);

  (void)in_sizes; (void)n_in; (void)out_size; (void)ws_size;
}
